// RNNRegressor_1872605741804
// MI455X (gfx1250) — matrix operand order repaired and checked by reading, not compiled, not run on hardware
//
#include <hip/hip_runtime.h>

typedef __attribute__((ext_vector_type(16))) _Float16 v16h;
typedef __attribute__((ext_vector_type(8)))  _Float16 v8h;
typedef __attribute__((ext_vector_type(8)))  float    v8f;

#define H   128
#define BSZ 256
#define T   2048
#define NW  (512*128)   // elements in one [4H,H] weight matrix

// d_out layout: y[256] | h_n[2][256][128] | c_n[2][256][128]
#define HN_OFF  256
#define CN_OFF  (256 + 2*BSZ*H)

// ---- fast activations: native v_tanh_f32 on gfx1250, rcp fallback (no IEEE div) ----
__device__ __forceinline__ float fast_tanh(float x) {
#if __has_builtin(__builtin_amdgcn_tanhf)
  return __builtin_amdgcn_tanhf(x);
#elif __has_builtin(__builtin_amdgcn_tanh_f32)
  return __builtin_amdgcn_tanh_f32(x);
#else
  float e = __expf(-2.0f * x);
  return (1.0f - e) * __builtin_amdgcn_rcpf(1.0f + e);
#endif
}
__device__ __forceinline__ float fast_sigmoid(float x) {
  return 0.5f + 0.5f * fast_tanh(0.5f * x);
}

// A-fragment (contiguous k order used here, same as the B loads): lane<16 -> K {kb..kb+15},
// lane>=16 -> K {kb+16..kb+31}; rowp = &data[row][0], contiguous in K.
__device__ __forceinline__ v16h load_a_frag(const _Float16* rowp, int kb, int half) {
  union { v16h v; v8h p[2]; } u;
  u.p[0] = *(const v8h*)(rowp + kb + 16 * half);
  u.p[1] = *(const v8h*)(rowp + kb + 8 + 16 * half);
  return u.v;
}

// -------------------- weight convert (f32 -> f16 staging in ws) --------------------
__global__ void convert_weights(const float* __restrict__ W0hh,
                                const float* __restrict__ W1ih,
                                const float* __restrict__ W1hh,
                                _Float16* __restrict__ ws) {
  int i = blockIdx.x * blockDim.x + threadIdx.x;
  if (i < NW)            ws[i] = (_Float16)W0hh[i];
  else if (i < 2 * NW)   ws[i] = (_Float16)W1ih[i - NW];
  else if (i < 3 * NW)   ws[i] = (_Float16)W1hh[i - 2 * NW];
}

// -------------------- layer 0: persistent scan over T --------------------
__global__ __launch_bounds__(256, 1)
void lstm_layer0(const float* __restrict__ x,       // [B,T] (IN==1)
                 const float* __restrict__ Wih,     // [512,1]
                 const float* __restrict__ bih,
                 const float* __restrict__ bhh,
                 const _Float16* __restrict__ Whh16,// [512][128] f16
                 _Float16* __restrict__ out0,       // ws [B][T][H] f16
                 float* __restrict__ dout) {
  __shared__ _Float16 h_lds[2][16][H];   // ping-pong h state
  __shared__ float    x_lds[2][16];      // ping-pong x stage

  const int tid  = threadIdx.x;
  const int wave = tid >> 5;
  const int lane = tid & 31;
  const int n    = lane & 15;
  const int half = lane >> 4;
  const int r0   = blockIdx.x * 16;
  const int col  = wave * 16 + n;

  float bias[4], wih[4];
#pragma unroll
  for (int g = 0; g < 4; ++g) {
    bias[g] = bih[g * H + col] + bhh[g * H + col];
    wih[g]  = Wih[g * H + col];
  }

  // B fragments (lanes 0-15: K 0-15; lanes 16-31: K 16-31; N = lane%16), pinned in VGPRs
  v16h Bw[4][4];
#pragma unroll
  for (int g = 0; g < 4; ++g)
#pragma unroll
    for (int kc = 0; kc < 4; ++kc)
      Bw[g][kc] = *(const v16h*)(Whh16 + (size_t)(g * H + col) * H + kc * 32 + half * 16);

  v8f c = {};
  float hv[8];
#pragma unroll
  for (int j = 0; j < 8; ++j) hv[j] = 0.0f;

  for (int i = tid; i < 2 * 16 * H; i += 256) ((_Float16*)h_lds)[i] = (_Float16)0.0f;
  if (tid < 16) x_lds[0][tid] = x[(size_t)(r0 + tid) * T];
  __syncthreads();

  for (int t = 0; t < T; ++t) {
    const int wb = t & 1;        // buffer written this step (h_t, read next step)
    const int rb = wb ^ 1;       // buffer holding h_{t-1}
    const _Float16* hrow = &h_lds[rb][n][0];

    v16h A[4];
#pragma unroll
    for (int kc = 0; kc < 4; ++kc) A[kc] = load_a_frag(hrow, kc * 32, half);
    float xv[8];
#pragma unroll
    for (int j = 0; j < 8; ++j) xv[j] = x_lds[wb][j + 8 * half];

    // gates = bias + x*Wih + h @ Whh^T  (16 WMMAs/wave/step)
    v8f acc[4];
#pragma unroll
    for (int g = 0; g < 4; ++g) {
      v8f a;
#pragma unroll
      for (int j = 0; j < 8; ++j) a[j] = bias[g] + xv[j] * wih[g];
#pragma unroll
      for (int kc = 0; kc < 4; ++kc)
        a = __builtin_amdgcn_wmma_f32_16x16x32_f16(false, A[kc], false, Bw[g][kc],
                                                   (short)0, a, false, false);
      acc[g] = a;
    }
#pragma unroll
    for (int j = 0; j < 8; ++j) {
      float iv = fast_sigmoid(acc[0][j]);
      float fv = fast_sigmoid(acc[1][j]);
      float gv = fast_tanh(acc[2][j]);
      float ov = fast_sigmoid(acc[3][j]);
      float cv = fv * c[j] + iv * gv;
      c[j]  = cv;
      hv[j] = ov * fast_tanh(cv);
    }

    // publish h_t (LDS ping buffer + out0 in L2-resident ws), stage x_{t+1}
#pragma unroll
    for (int j = 0; j < 8; ++j) {
      int row = j + 8 * half;
      _Float16 h16 = (_Float16)hv[j];
      h_lds[wb][row][col] = h16;
      out0[(size_t)(r0 + row) * T * H + (size_t)t * H + col] = h16;
    }
    if (tid < 16 && t + 1 < T) x_lds[wb ^ 1][tid] = x[(size_t)(r0 + tid) * T + (t + 1)];
    __syncthreads();   // single barrier per step (double-buffered state)
  }

#pragma unroll
  for (int j = 0; j < 8; ++j) {
    int row = j + 8 * half;
    dout[HN_OFF + (size_t)(r0 + row) * H + col] = hv[j];
    dout[CN_OFF + (size_t)(r0 + row) * H + col] = c[j];
  }
}

// -------------------- layer 1: fused input-projection + recurrence (K=256) --------------------
__global__ __launch_bounds__(256, 1)
void lstm_layer1(const _Float16* __restrict__ out0, // [B][T][H] f16
                 const float* __restrict__ bih,
                 const float* __restrict__ bhh,
                 const _Float16* __restrict__ Wih16,// [512][128] f16
                 const _Float16* __restrict__ Whh16,// [512][128] f16
                 float* __restrict__ dout) {
  __shared__ _Float16 h_lds[2][16][H];   // ping-pong h state

  const int tid  = threadIdx.x;
  const int wave = tid >> 5;
  const int lane = tid & 31;
  const int n    = lane & 15;
  const int half = lane >> 4;
  const int r0   = blockIdx.x * 16;
  const int col  = wave * 16 + n;

  float bias[4];
#pragma unroll
  for (int g = 0; g < 4; ++g) bias[g] = bih[g * H + col] + bhh[g * H + col];

  v16h Bi[4][4], Bh[4][4];
#pragma unroll
  for (int g = 0; g < 4; ++g)
#pragma unroll
    for (int kc = 0; kc < 4; ++kc) {
      Bi[g][kc] = *(const v16h*)(Wih16 + (size_t)(g * H + col) * H + kc * 32 + half * 16);
      Bh[g][kc] = *(const v16h*)(Whh16 + (size_t)(g * H + col) * H + kc * 32 + half * 16);
    }

  v8f c = {};
  float hv[8];
#pragma unroll
  for (int j = 0; j < 8; ++j) hv[j] = 0.0f;

  for (int i = tid; i < 2 * 16 * H; i += 256) ((_Float16*)h_lds)[i] = (_Float16)0.0f;
  __syncthreads();

  const _Float16* xrow = out0 + (size_t)(r0 + n) * T * H;

  for (int t = 0; t < T; ++t) {
    const int wb = t & 1;
    const int rb = wb ^ 1;
    const _Float16* hrow = &h_lds[rb][n][0];
    const _Float16* xt   = xrow + (size_t)t * H;

    v16h Ah[4], Ax[4];
#pragma unroll
    for (int kc = 0; kc < 4; ++kc) {
      Ax[kc] = load_a_frag(xt,   kc * 32, half);   // global (L2-resident out0)
      Ah[kc] = load_a_frag(hrow, kc * 32, half);   // LDS
    }
    if (t + 1 < T) __builtin_prefetch(xt + H, 0, 3);  // global_prefetch next timestep

    // gates = bias + h @ Whh^T + out0_t @ Wih^T  (32 WMMAs/wave/step)
    v8f acc[4];
#pragma unroll
    for (int g = 0; g < 4; ++g) {
      v8f a;
#pragma unroll
      for (int j = 0; j < 8; ++j) a[j] = bias[g];
#pragma unroll
      for (int kc = 0; kc < 4; ++kc)   // LDS h-path first: hides out0 load latency
        a = __builtin_amdgcn_wmma_f32_16x16x32_f16(false, Ah[kc], false, Bh[g][kc],
                                                   (short)0, a, false, false);
#pragma unroll
      for (int kc = 0; kc < 4; ++kc)
        a = __builtin_amdgcn_wmma_f32_16x16x32_f16(false, Ax[kc], false, Bi[g][kc],
                                                   (short)0, a, false, false);
      acc[g] = a;
    }
#pragma unroll
    for (int j = 0; j < 8; ++j) {
      float iv = fast_sigmoid(acc[0][j]);
      float fv = fast_sigmoid(acc[1][j]);
      float gv = fast_tanh(acc[2][j]);
      float ov = fast_sigmoid(acc[3][j]);
      float cv = fv * c[j] + iv * gv;
      c[j]  = cv;
      hv[j] = ov * fast_tanh(cv);
    }
#pragma unroll
    for (int j = 0; j < 8; ++j)
      h_lds[wb][j + 8 * half][col] = (_Float16)hv[j];
    __syncthreads();   // single barrier per step
  }

#pragma unroll
  for (int j = 0; j < 8; ++j) {
    int row = j + 8 * half;
    dout[HN_OFF + BSZ * H + (size_t)(r0 + row) * H + col] = hv[j];
    dout[CN_OFF + BSZ * H + (size_t)(r0 + row) * H + col] = c[j];
  }
}

// -------------------- final FC: y = h1_last @ Wfc^T + bfc --------------------
__global__ void fc_kernel(const float* __restrict__ Wfc, const float* __restrict__ bfc,
                          float* __restrict__ dout) {
  int b = blockIdx.x * blockDim.x + threadIdx.x;
  if (b < BSZ) {
    const float* h1 = dout + HN_OFF + BSZ * H + (size_t)b * H;
    float s = bfc[0];
#pragma unroll 8
    for (int k = 0; k < H; ++k) s += h1[k] * Wfc[k];
    dout[b] = s;
  }
}

extern "C" void kernel_launch(void* const* d_in, const int* in_sizes, int n_in,
                              void* d_out, int out_size, void* d_ws, size_t ws_size,
                              hipStream_t stream) {
  const float* x      = (const float*)d_in[0];
  const float* Wih0   = (const float*)d_in[1];
  const float* Whh0   = (const float*)d_in[2];
  const float* bih0   = (const float*)d_in[3];
  const float* bhh0   = (const float*)d_in[4];
  const float* Wih1   = (const float*)d_in[5];
  const float* Whh1   = (const float*)d_in[6];
  const float* bih1   = (const float*)d_in[7];
  const float* bhh1   = (const float*)d_in[8];
  const float* Wfc    = (const float*)d_in[9];
  const float* bfc    = (const float*)d_in[10];
  float* out          = (float*)d_out;

  _Float16* ws     = (_Float16*)d_ws;
  _Float16* W0hh16 = ws;
  _Float16* W1ih16 = ws + NW;
  _Float16* W1hh16 = ws + 2 * NW;
  _Float16* out0   = ws + 3 * NW;   // [B][T][H] f16, 128 MB, L2-resident

  convert_weights<<<(3 * NW + 255) / 256, 256, 0, stream>>>(Whh0, Wih1, Whh1, ws);
  lstm_layer0<<<BSZ / 16, 256, 0, stream>>>(x, Wih0, bih0, bhh0, W0hh16, out0, out);
  lstm_layer1<<<BSZ / 16, 256, 0, stream>>>(out0, bih1, bhh1, W1ih16, W1hh16, out);
  fc_kernel<<<1, 256, 0, stream>>>(Wfc, bfc, out);
}
